// Attention_21698174779835
// MI455X (gfx1250) — compile-verified, matrix operand order checked by reading, not run on hardware
//
#include <hip/hip_runtime.h>

// ---------------- problem constants ----------------
#define B_    16
#define TQ    2048
#define TK    2048
#define DIM   256
#define MT    32            // q rows per workgroup
#define SCALE 0.0625f       // 1/sqrt(256)

// ---------------- LDS layout (bytes) ----------------
#define SQ_STRIDE 264       // halves per Q row (padded)
#define SS_STRIDE 2052      // floats per score row (padded: 2052%64==4 -> conflict-free column reads)
#define SV_STRIDE 260       // floats per V row (padded; 64 16B-chunks per row stay within row)

#define SMEM_SQ_OFF   0
#define SMEM_SS_OFF   (MT * SQ_STRIDE * 2)                 //  16,896
#define SMEM_SV_OFF   (SMEM_SS_OFF + MT * SS_STRIDE * 4)   // 279,552
#define SMEM_MAX_OFF  (SMEM_SV_OFF + 32 * SV_STRIDE * 4)   // 312,832
#define SMEM_INV_OFF  (SMEM_MAX_OFF + MT * 4)
#define SMEM_PART_OFF (SMEM_INV_OFF + MT * 4)
#define SMEM_BYTES    (SMEM_PART_OFF + MT * 8 * 4)         // 314,112 < 320KB

typedef __attribute__((ext_vector_type(16))) _Float16 v16h;
typedef __attribute__((ext_vector_type(8)))  _Float16 v8h;
typedef __attribute__((ext_vector_type(8)))  float    v8f;

union V16H { v16h v; v8h p[2]; _Float16 h[16]; };
union F4x2 { float4 f4[2]; float f[8]; };

__global__ __launch_bounds__(256, 1)
void attn_fused_wmma(const float* __restrict__ K,
                     const float* __restrict__ V,
                     const float* __restrict__ Q,
                     float* __restrict__ outR,   // [B, TQ, 2*DIM]
                     float* __restrict__ outA)   // [B, TQ, TK]
{
    extern __shared__ char smem[];
    _Float16* sQ   = (_Float16*)(smem + SMEM_SQ_OFF);
    float*    sS   = (float*)   (smem + SMEM_SS_OFF);
    float*    sV   = (float*)   (smem + SMEM_SV_OFF);
    float*    sMax = (float*)   (smem + SMEM_MAX_OFF);
    float*    sInv = (float*)   (smem + SMEM_INV_OFF);
    float*    sPart= (float*)   (smem + SMEM_PART_OFF);

    const int tid  = threadIdx.x;
    const int lane = tid & 31;          // wave32
    const int wave = tid >> 5;          // 0..7
    const int l15  = lane & 15;
    const int kh   = lane >> 4;         // K-half selector per ISA 16-bit A/B layout

    const int b  = blockIdx.y;
    const int q0 = blockIdx.x * MT;

    const float* Kb = K + (size_t)b * TK * DIM;
    const float* Vb = V + (size_t)b * TK * DIM;
    const float* Qb = Q + (size_t)b * TQ * DIM;
    float* Rb = outR + (size_t)b * TQ * (2 * DIM);
    float* Ab = outA + (size_t)b * TQ * TK;

    // ---------- Phase 0: stage Q tile -> LDS f16; copy Q into R[:, DIM:2*DIM] ----------
    #pragma unroll 1
    for (int i = 0; i < (MT * DIM) / 256; ++i) {
        int flat = i * 256 + tid;
        int r = flat >> 8, c = flat & (DIM - 1);
        float q = Qb[(size_t)(q0 + r) * DIM + c];
        sQ[r * SQ_STRIDE + c] = (_Float16)q;
        Rb[(size_t)(q0 + r) * (2 * DIM) + DIM + c] = q;
    }
    __syncthreads();

    // ---------- Phase 1: S = scale * Q @ K^T  (each wave: disjoint 16x16 subtile per n-step) ----------
    const int mi = wave >> 2;           // 0..1  (m subtile)
    const int ni = wave & 3;            // 0..3  (n subtile within 64-col step)
    const _Float16* qrow = sQ + (16 * mi + l15) * SQ_STRIDE;

    #pragma unroll 1
    for (int nstep = 0; nstep < TK / 64; ++nstep) {
        const int ncol = nstep * 64 + 16 * ni + l15;      // K row this lane supplies (B operand col)
        const float* krow = Kb + (size_t)ncol * DIM;
        v8f c = {0.f, 0.f, 0.f, 0.f, 0.f, 0.f, 0.f, 0.f};
        #pragma unroll
        for (int ks = 0; ks < DIM / 32; ++ks) {
            const int kb = ks * 32 + kh * 8;
            V16H a, bb;
            a.p[0] = *(const v8h*)(qrow + kb);            // K = khalf*8 + 0..7
            a.p[1] = *(const v8h*)(qrow + kb + 16);       // K = 16 + khalf*8 + 0..7
            F4x2 lo, hi;
            lo.f4[0] = *(const float4*)(krow + kb);
            lo.f4[1] = *(const float4*)(krow + kb + 4);
            hi.f4[0] = *(const float4*)(krow + kb + 16);
            hi.f4[1] = *(const float4*)(krow + kb + 20);
            #pragma unroll
            for (int j = 0; j < 8; ++j) {
                bb.h[j]     = (_Float16)lo.f[j];
                bb.h[8 + j] = (_Float16)hi.f[j];
            }
            c = __builtin_amdgcn_wmma_f32_16x16x32_f16(
                    false, a.v, false, bb.v, (short)0, c, false, false);
        }
        // C layout: VGPR r -> M = r + 8*kh ; lane[3:0] -> N
        const int crow = 16 * mi + 8 * kh;
        const int ccol = nstep * 64 + 16 * ni + l15;
        #pragma unroll
        for (int r = 0; r < 8; ++r)
            sS[(crow + r) * SS_STRIDE + ccol] = c[r] * SCALE;
    }
    __syncthreads();

    // ---------- Phase 2: exact softmax over 32x2048 strip; write A; keep P (fp32) in sS ----------
    {
        const int row = tid >> 3, t8 = tid & 7;           // 8 threads per row
        const float* srow = sS + row * SS_STRIDE + t8 * (TK / 8);
        float m = -3.0e38f;
        #pragma unroll 1
        for (int i = 0; i < TK / 8; ++i) m = fmaxf(m, srow[i]);
        sPart[row * 8 + t8] = m;
        __syncthreads();
        if (t8 == 0) {
            float mm = sPart[row * 8];
            #pragma unroll
            for (int j = 1; j < 8; ++j) mm = fmaxf(mm, sPart[row * 8 + j]);
            sMax[row] = mm;
        }
        __syncthreads();
        const float rm = sMax[row];
        float s = 0.f;
        #pragma unroll 1
        for (int i = 0; i < TK / 8; ++i) s += __expf(srow[i] - rm);
        sPart[row * 8 + t8] = s;
        __syncthreads();
        if (t8 == 0) {
            float ss = 0.f;
            #pragma unroll
            for (int j = 0; j < 8; ++j) ss += sPart[row * 8 + j];
            sInv[row] = 1.0f / ss;
        }
        __syncthreads();
        // coalesced normalize + store A (consecutive tid -> consecutive k)
        #pragma unroll 1
        for (int i = 0; i < (MT * TK) / 256; ++i) {
            int flat = i * 256 + tid;
            int r = flat >> 11, k = flat & (TK - 1);
            float p = __expf(sS[r * SS_STRIDE + k] - sMax[r]) * sInv[r];
            sS[r * SS_STRIDE + k] = p;
            Ab[(size_t)(q0 + r) * TK + k] = p;
        }
    }

    // ---------- Phase 3: R = P @ V  (wave owns 32 output cols; k-steps of 32) ----------
    // V tiles staged with GLOBAL_LOAD_ASYNC_TO_LDS_B128 (ASYNCcnt): no VGPR data
    // movement, LDS written directly by the memory system. Dynamic LDS starts at
    // offset 0 (no static __shared__), so LDS destinations are plain byte offsets.
    v8f acc[2][2] = {};
    const int d0 = wave * 32;
    #pragma unroll 1
    for (int kt = 0; kt < TK / 32; ++kt) {
        const int k0 = kt * 32;
        __syncthreads();                                  // prior tile fully consumed
        if (k0 + 32 < TK)
            __builtin_prefetch(Vb + (size_t)(k0 + 32) * DIM + tid, 0, 1);
        #pragma unroll
        for (int i = 0; i < 8; ++i) {                     // 8 x (256 lanes x 16B) = 32KB tile
            int chunk = i * 256 + tid;                    // 16-byte chunk id
            int r  = chunk >> 6;                          // 64 chunks per 256-float row
            int c4 = (chunk & 63) << 2;                   // starting float column
            unsigned lds_off = (unsigned)(SMEM_SV_OFF + (r * SV_STRIDE + c4) * 4);
            unsigned long long ga =
                (unsigned long long)(const void*)(Vb + (size_t)(k0 + r) * DIM + c4);
            asm volatile("global_load_async_to_lds_b128 %0, %1, off"
                         :: "v"(lds_off), "v"(ga)
                         : "memory");
        }
        asm volatile("s_wait_asynccnt 0" ::: "memory");
        __syncthreads();

        V16H a0, a1, b0, b1;
        const float* p0 = sS + (l15)      * SS_STRIDE + k0;
        const float* p1 = sS + (16 + l15) * SS_STRIDE + k0;
        #pragma unroll
        for (int j = 0; j < 8; ++j) {
            a0.h[j]     = (_Float16)p0[kh * 8 + j];
            a0.h[8 + j] = (_Float16)p0[16 + kh * 8 + j];
            a1.h[j]     = (_Float16)p1[kh * 8 + j];
            a1.h[8 + j] = (_Float16)p1[16 + kh * 8 + j];
            b0.h[j]     = (_Float16)sV[(kh * 8 + j)      * SV_STRIDE + d0 + l15];
            b0.h[8 + j] = (_Float16)sV[(16 + kh * 8 + j) * SV_STRIDE + d0 + l15];
            b1.h[j]     = (_Float16)sV[(kh * 8 + j)      * SV_STRIDE + d0 + 16 + l15];
            b1.h[8 + j] = (_Float16)sV[(16 + kh * 8 + j) * SV_STRIDE + d0 + 16 + l15];
        }
        acc[0][0] = __builtin_amdgcn_wmma_f32_16x16x32_f16(false, a0.v, false, b0.v, (short)0, acc[0][0], false, false);
        acc[0][1] = __builtin_amdgcn_wmma_f32_16x16x32_f16(false, a0.v, false, b1.v, (short)0, acc[0][1], false, false);
        acc[1][0] = __builtin_amdgcn_wmma_f32_16x16x32_f16(false, a1.v, false, b0.v, (short)0, acc[1][0], false, false);
        acc[1][1] = __builtin_amdgcn_wmma_f32_16x16x32_f16(false, a1.v, false, b1.v, (short)0, acc[1][1], false, false);
    }

    // ---------- Epilogue: write attention output into R[:, 0:DIM] ----------
    #pragma unroll
    for (int m2 = 0; m2 < 2; ++m2)
        #pragma unroll
        for (int nj = 0; nj < 2; ++nj) {
            const int col = d0 + 16 * nj + l15;
            #pragma unroll
            for (int r = 0; r < 8; ++r) {
                const int rowi = 16 * m2 + 8 * kh + r;
                Rb[(size_t)(q0 + rowi) * (2 * DIM) + col] = acc[m2][nj][r];
            }
        }
}

extern "C" void kernel_launch(void* const* d_in, const int* in_sizes, int n_in,
                              void* d_out, int out_size, void* d_ws, size_t ws_size,
                              hipStream_t stream)
{
    (void)in_sizes; (void)n_in; (void)out_size; (void)d_ws; (void)ws_size;
    const float* K = (const float*)d_in[0];
    const float* V = (const float*)d_in[1];
    const float* Q = (const float*)d_in[2];
    float* outR = (float*)d_out;                                  // B*TQ*2*DIM floats
    float* outA = outR + (size_t)B_ * TQ * (2 * DIM);             // B*TQ*TK floats

    hipFuncSetAttribute((const void*)attn_fused_wmma,
                        hipFuncAttributeMaxDynamicSharedMemorySize, SMEM_BYTES);

    dim3 grid(TQ / MT, B_);
    attn_fused_wmma<<<grid, 256, SMEM_BYTES, stream>>>(K, V, Q, outR, outA);
}
